// PointNet4DTemporalFusion_68436008895054
// MI455X (gfx1250) — matrix operand order repaired and checked by reading, not compiled, not run on hardware
//
#include <hip/hip_runtime.h>
#include <math.h>

// ---------------- model constants ----------------
#define DIMM     1024
#define DEPTH    5
#define HEADS    4
#define DIM_HEAD 1024
#define INNERD   4096           // HEADS*DIM_HEAD
#define MLP_D    4096
#define BB       8
#define TT       150
#define D_IN     2048           // 2*DIM
#define D_STATE  16
#define D_CONV   4
#define DT_RANK  64
#define LN_EPSF  1e-5f
#define MTOK     (BB*TT)        // 1200 rows
#define QKVLD    (3*INNERD)     // 12288
#define TP       160            // T padded to 16

typedef __attribute__((ext_vector_type(16))) _Float16 v16h;
typedef __attribute__((ext_vector_type(8)))  float    v8f;
typedef __fp16 fp16x2 __attribute__((ext_vector_type(2)));

__device__ __forceinline__ float sigmoidf_(float x){ return 1.f/(1.f+__expf(-x)); }
__device__ __forceinline__ float siluf_(float x){ return x*sigmoidf_(x); }
__device__ __forceinline__ float geluf_(float x){ return 0.5f*x*(1.f+erff(x*0.70710678118f)); }
__device__ __forceinline__ float softplusf_(float x){ return (x>20.f)?x:log1pf(__expf(x)); }
__device__ __forceinline__ float applyact(float v, int act){
  if (act==1) return siluf_(v);
  if (act==2) return geluf_(v);
  if (act==3) return softplusf_(v);
  return v;
}
__device__ __forceinline__ unsigned pk2(float a, float b){
  union { fp16x2 h; unsigned u; } c; c.h = __builtin_amdgcn_cvt_pkrtz(a, b); return c.u;
}

// =====================================================================
// WMMA GEMM: f16 compute / f32 accumulate, software-pipelined staging,
// double-buffered LDS (one barrier per k-chunk), branch-free guards.
//   transB=1:  C[M,N] = act( scale * A[M,K] @ W[N,K]^T + bias[N] )
//              (K % 32 == 0 required; n rows clamped+masked)
//   transB=0:  C[M,N] = act( scale * A[M,K] @ W[K,N]   + bias[N] )
//              (N % 4 == 0; W rows up to 32*ceil(K/32) must be readable;
//               A cols [K, 32*ceil(K/32)) must be ZERO — softmax pads them)
// A rows beyond M must be readable (workspace layout guarantees it);
// garbage rows only feed output rows that are never stored.
// block = 256 threads = 8 waves; tile = 32(M) x 128(N); each wave owns two
// 16x16 C tiles sharing one B fragment -> 2 WMMA per barrier.
// =====================================================================
__global__ __launch_bounds__(256)
void k_gemm(const float* __restrict__ A, int lda,
            const float* __restrict__ W, int ldw,
            float* __restrict__ C, int ldc,
            const float* __restrict__ bias,
            int M, int N, int K, int act, float scale, int transB)
{
  __shared__ unsigned As[2][32*16];    // 32 m  x 32 k  (f16 pairs), x2 buffers
  __shared__ unsigned Ws[2][128*16];   // 128 n x 32 k  (f16 pairs), x2 buffers
  const int tid  = threadIdx.x;
  const int lane = tid & 31;
  const int wave = tid >> 5;
  const int half = lane >> 4;
  const int m0 = blockIdx.y * 32;
  const int n0 = blockIdx.x * 128;
  const int ktiles = (K + 31) >> 5;

  // ---- loop-invariant per-thread source pointers & masks ----
  const int ar = tid >> 3, akg = tid & 7;           // A/W(T): row, 4-col group
  const float* pA = A + (size_t)(m0 + ar)*lda + akg*4;

  const float* pW0; const float* pW1; const float* pW2; const float* pW3;
  int mk0 = 1, mk1 = 1, mk2 = 1, mk3 = 1;
  size_t wstep;
  if (transB) {
    int n_0 = n0 + 0*32 + ar, n_1 = n0 + 1*32 + ar, n_2 = n0 + 2*32 + ar, n_3 = n0 + 3*32 + ar;
    mk0 = n_0 < N; mk1 = n_1 < N; mk2 = n_2 < N; mk3 = n_3 < N;
    pW0 = W + (size_t)(mk0 ? n_0 : N-1)*ldw + akg*4;
    pW1 = W + (size_t)(mk1 ? n_1 : N-1)*ldw + akg*4;
    pW2 = W + (size_t)(mk2 ? n_2 : N-1)*ldw + akg*4;
    pW3 = W + (size_t)(mk3 ? n_3 : N-1)*ldw + akg*4;
    wstep = 32;
  } else {
    int kk = tid >> 5, ng = tid & 31;               // k row, 4-wide n group
    pW0 = W + (size_t)(kk +  0)*ldw + n0 + ng*4;
    pW1 = W + (size_t)(kk +  8)*ldw + n0 + ng*4;
    pW2 = W + (size_t)(kk + 16)*ldw + n0 + ng*4;
    pW3 = W + (size_t)(kk + 24)*ldw + n0 + ng*4;
    wstep = (size_t)32*ldw;
  }

  // ---- prologue: first chunk global -> regs ----
  float4 ra  = *(const float4*)pA;
  float4 rw0 = *(const float4*)pW0;
  float4 rw1 = *(const float4*)pW1;
  float4 rw2 = *(const float4*)pW2;
  float4 rw3 = *(const float4*)pW3;

  v8f acc0 = {}, acc1 = {};
  int buf = 0;

  for (int kt = 0; kt < ktiles; ++kt) {
    // ---- regs -> LDS (convert f32 -> packed f16) ----
    As[buf][ar*16 + akg*2 + 0] = pk2(ra.x, ra.y);
    As[buf][ar*16 + akg*2 + 1] = pk2(ra.z, ra.w);
    if (transB) {
      Ws[buf][( 0 + ar)*16 + akg*2 + 0] = mk0 ? pk2(rw0.x, rw0.y) : 0u;
      Ws[buf][( 0 + ar)*16 + akg*2 + 1] = mk0 ? pk2(rw0.z, rw0.w) : 0u;
      Ws[buf][(32 + ar)*16 + akg*2 + 0] = mk1 ? pk2(rw1.x, rw1.y) : 0u;
      Ws[buf][(32 + ar)*16 + akg*2 + 1] = mk1 ? pk2(rw1.z, rw1.w) : 0u;
      Ws[buf][(64 + ar)*16 + akg*2 + 0] = mk2 ? pk2(rw2.x, rw2.y) : 0u;
      Ws[buf][(64 + ar)*16 + akg*2 + 1] = mk2 ? pk2(rw2.z, rw2.w) : 0u;
      Ws[buf][(96 + ar)*16 + akg*2 + 0] = mk3 ? pk2(rw3.x, rw3.y) : 0u;
      Ws[buf][(96 + ar)*16 + akg*2 + 1] = mk3 ? pk2(rw3.z, rw3.w) : 0u;
    } else {
      _Float16* Wh = (_Float16*)Ws[buf];
      const int kk = tid >> 5, ng = tid & 31;
      Wh[(ng*4+0)*32 + kk +  0] = (_Float16)rw0.x;
      Wh[(ng*4+1)*32 + kk +  0] = (_Float16)rw0.y;
      Wh[(ng*4+2)*32 + kk +  0] = (_Float16)rw0.z;
      Wh[(ng*4+3)*32 + kk +  0] = (_Float16)rw0.w;
      Wh[(ng*4+0)*32 + kk +  8] = (_Float16)rw1.x;
      Wh[(ng*4+1)*32 + kk +  8] = (_Float16)rw1.y;
      Wh[(ng*4+2)*32 + kk +  8] = (_Float16)rw1.z;
      Wh[(ng*4+3)*32 + kk +  8] = (_Float16)rw1.w;
      Wh[(ng*4+0)*32 + kk + 16] = (_Float16)rw2.x;
      Wh[(ng*4+1)*32 + kk + 16] = (_Float16)rw2.y;
      Wh[(ng*4+2)*32 + kk + 16] = (_Float16)rw2.z;
      Wh[(ng*4+3)*32 + kk + 16] = (_Float16)rw2.w;
      Wh[(ng*4+0)*32 + kk + 24] = (_Float16)rw3.x;
      Wh[(ng*4+1)*32 + kk + 24] = (_Float16)rw3.y;
      Wh[(ng*4+2)*32 + kk + 24] = (_Float16)rw3.z;
      Wh[(ng*4+3)*32 + kk + 24] = (_Float16)rw3.w;
    }

    // ---- prefetch next chunk (issued before barrier; overlaps WMMA) ----
    if (kt + 1 < ktiles) {
      pA += 32;     ra  = *(const float4*)pA;
      pW0 += wstep; rw0 = *(const float4*)pW0;
      pW1 += wstep; rw1 = *(const float4*)pW1;
      pW2 += wstep; rw2 = *(const float4*)pW2;
      pW3 += wstep; rw3 = *(const float4*)pW3;
    }

    __syncthreads();   // single barrier per chunk (double-buffered LDS)

    // ---- fragments per ISA 16-bit layouts ----
    union { unsigned u[8]; v16h v; } af0, af1, bf;
    const int ml = lane & 15;
    #pragma unroll
    for (int r = 0; r < 8; ++r) {
      int p = ((r & 4) << 1) + half*4 + (r & 3);    // A pair index
      af0.u[r] = As[buf][ ml      *16 + p];
      af1.u[r] = As[buf][(ml + 16)*16 + p];
    }
    const int nl = (wave << 4) + ml;
    #pragma unroll
    for (int r = 0; r < 8; ++r)
      bf.u[r] = Ws[buf][nl*16 + ((r & 4) << 1) + half*4 + (r & 3)];        // B: same k order as A (table order)

    acc0 = __builtin_amdgcn_wmma_f32_16x16x32_f16(false, af0.v, false, bf.v, (short)0, acc0, false, false);
    acc1 = __builtin_amdgcn_wmma_f32_16x16x32_f16(false, af1.v, false, bf.v, (short)0, acc1, false, false);
    buf ^= 1;
  }

  // ---- epilogue: VGPR r -> M = r + 8*half, N = lane%16 ----
  const int n = n0 + (wave << 4) + (lane & 15);
  if (n < N) {
    const float bv = bias ? bias[n] : 0.f;
    #pragma unroll
    for (int r = 0; r < 8; ++r) {
      int m = m0 + r + half*8;
      if (m < M)      C[(size_t)m*ldc + n]      = applyact(acc0[r]*scale + bv, act);
      if (m + 16 < M) C[(size_t)(m+16)*ldc + n] = applyact(acc1[r]*scale + bv, act);
    }
  }
}

// =====================================================================
// LayerNorm over DIM=1024 (one block per token, LDS tree reduce)
// =====================================================================
__global__ __launch_bounds__(256)
void k_layernorm(const float* __restrict__ x, const float* __restrict__ w,
                 const float* __restrict__ b, float* __restrict__ out)
{
  __shared__ float s1[256], s2[256];
  const int row = blockIdx.x;
  const float* xr = x + (size_t)row*DIMM;
  float v[4]; float sum = 0.f, sq = 0.f;
  #pragma unroll
  for (int j = 0; j < 4; ++j){ v[j] = xr[threadIdx.x + j*256]; sum += v[j]; sq += v[j]*v[j]; }
  s1[threadIdx.x] = sum; s2[threadIdx.x] = sq; __syncthreads();
  for (int off = 128; off > 0; off >>= 1){
    if (threadIdx.x < off){ s1[threadIdx.x] += s1[threadIdx.x+off]; s2[threadIdx.x] += s2[threadIdx.x+off]; }
    __syncthreads();
  }
  const float mu   = s1[0] * (1.f/DIMM);
  const float var  = s2[0] * (1.f/DIMM) - mu*mu;
  const float rstd = rsqrtf(var + LN_EPSF);
  #pragma unroll
  for (int j = 0; j < 4; ++j){
    int c = threadIdx.x + j*256;
    out[(size_t)row*DIMM + c] = (v[j]-mu)*rstd*w[c] + b[c];
  }
}

// =====================================================================
// Causal depthwise conv (4 taps) + bias + SiLU (reads xc half of xz)
// =====================================================================
__global__ void k_conv_silu(const float* __restrict__ xz, const float* __restrict__ cw,
                            const float* __restrict__ cb, float* __restrict__ xc)
{
  int idx = blockIdx.x*blockDim.x + threadIdx.x;
  if (idx >= MTOK*D_IN) return;
  int d  = idx & (D_IN-1);
  int bt = idx >> 11;
  int t = bt % TT, b = bt / TT;
  float acc = cb[d];
  #pragma unroll
  for (int j = 0; j < D_CONV; ++j) {
    int tt = t - (D_CONV-1) + j;
    if (tt >= 0) acc += xz[(size_t)(b*TT+tt)*(2*D_IN) + d] * cw[d*D_CONV + j];
  }
  xc[idx] = siluf_(acc);
}

// =====================================================================
// Selective scan: one thread per (b, channel); 16 states in registers.
// =====================================================================
__global__ __launch_bounds__(256)
void k_scan(const float* __restrict__ dt, const float* __restrict__ xc,
            const float* __restrict__ xdbl, const float* __restrict__ xz,
            const float* __restrict__ Alog, const float* __restrict__ Dskip,
            float* __restrict__ y)
{
  int idx = blockIdx.x*blockDim.x + threadIdx.x;     // BB*D_IN = 16384 exact
  int d = idx & (D_IN-1), b = idx >> 11;
  float An[D_STATE], h[D_STATE];
  #pragma unroll
  for (int s = 0; s < D_STATE; ++s){ An[s] = -__expf(Alog[(size_t)d*D_STATE + s]); h[s] = 0.f; }
  const float Dv = Dskip[d];
  for (int t = 0; t < TT; ++t) {
    size_t rt = (size_t)(b*TT + t);
    float dtv = dt[rt*D_IN + d];
    float xv  = xc[rt*D_IN + d];
    float zv  = xz[rt*(2*D_IN) + D_IN + d];
    const float* row = xdbl + rt*(DT_RANK + 2*D_STATE);
    float yv = 0.f;
    #pragma unroll
    for (int s = 0; s < D_STATE; ++s) {
      float dA = __expf(dtv*An[s]);
      h[s] = dA*h[s] + dtv*row[DT_RANK + s]*xv;
      yv  += h[s]*row[DT_RANK + D_STATE + s];
    }
    y[rt*D_IN + d] = (yv + Dv*xv) * siluf_(zv);
  }
}

// =====================================================================
// Causal softmax over one row; zero-pads cols [TT, TP) so the padded-K
// attn@V GEMM accumulates exact zeros for j >= TT.
// =====================================================================
__global__ __launch_bounds__(256)
void k_softmax_causal(float* __restrict__ dots)
{
  __shared__ float red[256];
  const int z = blockIdx.x;                 // 0 .. B*H*T-1
  const int i = z % TT;
  float* row = dots + (size_t)(z / TT)*TP*TP + (size_t)i*TP;
  const int tid = threadIdx.x;
  float v = -3.4e38f;
  if (tid < TT && tid <= i) v = row[tid];
  red[tid] = v; __syncthreads();
  for (int off = 128; off > 0; off >>= 1){ if (tid < off) red[tid] = fmaxf(red[tid], red[tid+off]); __syncthreads(); }
  const float mx = red[0]; __syncthreads();
  float p = 0.f;
  if (tid < TT && tid <= i) p = __expf(v - mx);
  red[tid] = p; __syncthreads();
  for (int off = 128; off > 0; off >>= 1){ if (tid < off) red[tid] += red[tid+off]; __syncthreads(); }
  const float inv = 1.f/red[0];
  if (tid < TP) row[tid] = (tid < TT) ? p*inv : 0.f;
}

__global__ void k_add(float* __restrict__ x, const float* __restrict__ t, int n){
  int i = blockIdx.x*blockDim.x + threadIdx.x;
  if (i < n) x[i] += t[i];
}
__global__ void k_copy(float* __restrict__ o, const float* __restrict__ s, int n){
  int i = blockIdx.x*blockDim.x + threadIdx.x;
  if (i < n) o[i] = s[i];
}

// ---------------- host-side launch helper ----------------
static inline void gemm(hipStream_t s, const float* A, int lda, const float* W, int ldw,
                        float* C, int ldc, int M, int N, int K,
                        const float* bias, int act, float scale, int transB)
{
  dim3 g((N + 127)/128, (M + 31)/32, 1);
  k_gemm<<<g, 256, 0, s>>>(A, lda, W, ldw, C, ldc, bias, M, N, K, act, scale, transB);
}

extern "C" void kernel_launch(void* const* d_in, const int* in_sizes, int n_in,
                              void* d_out, int out_size, void* d_ws, size_t ws_size,
                              hipStream_t stream)
{
  (void)in_sizes; (void)n_in; (void)out_size; (void)ws_size;
  const float* features   = (const float*)d_in[0];
  const float* mnorm_w    = (const float*)d_in[1];
  const float* mnorm_b    = (const float*)d_in[2];
  const float* in_proj_w  = (const float*)d_in[3];
  const float* conv_w     = (const float*)d_in[4];
  const float* conv_b     = (const float*)d_in[5];
  const float* x_proj_w   = (const float*)d_in[6];
  const float* dt_proj_w  = (const float*)d_in[7];
  const float* dt_proj_b  = (const float*)d_in[8];
  const float* A_log      = (const float*)d_in[9];
  const float* D_skip     = (const float*)d_in[10];
  const float* mamba_ow   = (const float*)d_in[11];
  const float* ln1_w      = (const float*)d_in[12];
  const float* ln1_b      = (const float*)d_in[13];
  const float* qkv_w      = (const float*)d_in[14];
  const float* attn_ow    = (const float*)d_in[15];
  const float* attn_ob    = (const float*)d_in[16];
  const float* ln2_w      = (const float*)d_in[17];
  const float* ln2_b      = (const float*)d_in[18];
  const float* ffn_w1     = (const float*)d_in[19];
  const float* ffn_b1     = (const float*)d_in[20];
  const float* ffn_w2     = (const float*)d_in[21];
  const float* ffn_b2     = (const float*)d_in[22];

  float* x  = (float*)d_out;                    // running residual state
  float* ws = (float*)d_ws;

  // workspace arena. Ordering guarantees that the GEMM's unguarded A-tile
  // over-reads (<= 31 rows past M) always land inside d_ws.
  float* xn    = ws;                                    // 1200*1024
  float* tmp   = ws + (size_t)MTOK*DIMM;                // 1200*1024
  float* arena = ws + (size_t)2*MTOK*DIMM;
  // mamba phase
  float* xz   = arena;                                  // 1200*4096
  float* xc   = xz   + (size_t)MTOK*2*D_IN;             // 1200*2048
  float* xdbl = xc   + (size_t)MTOK*D_IN;               // 1200*96
  float* dt   = xdbl + (size_t)MTOK*(DT_RANK+2*D_STATE);// 1200*2048
  float* ybuf = dt   + (size_t)MTOK*D_IN;               // 1200*2048
  // attention phase (reuses arena after mamba)
  float* qkv  = arena;                                  // 1200*12288
  float* ctx  = qkv  + (size_t)MTOK*QKVLD;              // 1200*4096 (before dots!)
  float* dots = ctx  + (size_t)MTOK*INNERD;             // 32*160*160 (rows<160 only)
  // ffn phase
  float* hid  = arena;                                  // 1200*4096

  const int NX = MTOK*DIMM;
  k_copy<<<(NX+255)/256, 256, 0, stream>>>(x, features, NX);

  for (int L = 0; L < DEPTH; ++L) {
    // ------------- Mamba block -------------
    k_layernorm<<<MTOK, 256, 0, stream>>>(x, mnorm_w + L*DIMM, mnorm_b + L*DIMM, xn);
    gemm(stream, xn, DIMM, in_proj_w + (size_t)L*2*D_IN*DIMM, DIMM,
         xz, 2*D_IN, MTOK, 2*D_IN, DIMM, nullptr, 0, 1.f, 1);
    k_conv_silu<<<(MTOK*D_IN+255)/256, 256, 0, stream>>>(
        xz, conv_w + (size_t)L*D_IN*D_CONV, conv_b + (size_t)L*D_IN, xc);
    gemm(stream, xc, D_IN, x_proj_w + (size_t)L*(DT_RANK+2*D_STATE)*D_IN, D_IN,
         xdbl, DT_RANK+2*D_STATE, MTOK, DT_RANK+2*D_STATE, D_IN, nullptr, 0, 1.f, 1);
    gemm(stream, xdbl, DT_RANK+2*D_STATE, dt_proj_w + (size_t)L*D_IN*DT_RANK, DT_RANK,
         dt, D_IN, MTOK, D_IN, DT_RANK, dt_proj_b + (size_t)L*D_IN, 3, 1.f, 1);
    k_scan<<<(BB*D_IN)/256, 256, 0, stream>>>(
        dt, xc, xdbl, xz, A_log + (size_t)L*D_IN*D_STATE, D_skip + (size_t)L*D_IN, ybuf);
    gemm(stream, ybuf, D_IN, mamba_ow + (size_t)L*DIMM*D_IN, D_IN,
         tmp, DIMM, MTOK, DIMM, D_IN, nullptr, 0, 1.f, 1);
    k_add<<<(NX+255)/256, 256, 0, stream>>>(x, tmp, NX);

    // ------------- Attention block -------------
    k_layernorm<<<MTOK, 256, 0, stream>>>(x, ln1_w + L*DIMM, ln1_b + L*DIMM, xn);
    gemm(stream, xn, DIMM, qkv_w + (size_t)L*QKVLD*DIMM, DIMM,
         qkv, QKVLD, MTOK, QKVLD, DIMM, nullptr, 0, 1.f, 1);
    for (int z = 0; z < BB*HEADS; ++z) {          // per-(b,h) q@k^T
      int b = z/HEADS, h = z%HEADS;
      const float* q = qkv + (size_t)b*TT*QKVLD + (size_t)h*DIM_HEAD;
      const float* k = q + INNERD;
      gemm(stream, q, QKVLD, k, QKVLD, dots + (size_t)z*TP*TP, TP,
           TT, TT, DIM_HEAD, nullptr, 0, 0.03125f /* 1/sqrt(1024) */, 1);
    }
    k_softmax_causal<<<BB*HEADS*TT, 256, 0, stream>>>(dots);
    for (int z = 0; z < BB*HEADS; ++z) {          // per-(b,h) attn @ v (NN)
      int b = z/HEADS, h = z%HEADS;
      const float* v = qkv + (size_t)b*TT*QKVLD + (size_t)2*INNERD + (size_t)h*DIM_HEAD;
      gemm(stream, dots + (size_t)z*TP*TP, TP, v, QKVLD,
           ctx + (size_t)b*TT*INNERD + (size_t)h*DIM_HEAD, INNERD,
           TT, DIM_HEAD, TT, nullptr, 0, 1.f, 0);
    }
    gemm(stream, ctx, INNERD, attn_ow + (size_t)L*DIMM*INNERD, INNERD,
         tmp, DIMM, MTOK, DIMM, INNERD, attn_ob + (size_t)L*DIMM, 2 /*gelu*/, 1.f, 1);
    k_add<<<(NX+255)/256, 256, 0, stream>>>(x, tmp, NX);

    // ------------- FFN block -------------
    k_layernorm<<<MTOK, 256, 0, stream>>>(x, ln2_w + L*DIMM, ln2_b + L*DIMM, xn);
    gemm(stream, xn, DIMM, ffn_w1 + (size_t)L*MLP_D*DIMM, DIMM,
         hid, MLP_D, MTOK, MLP_D, DIMM, ffn_b1 + (size_t)L*MLP_D, 2 /*gelu*/, 1.f, 1);
    gemm(stream, hid, MLP_D, ffn_w2 + (size_t)L*DIMM*MLP_D, MLP_D,
         tmp, DIMM, MTOK, DIMM, MLP_D, ffn_b2 + (size_t)L*DIMM, 0, 1.f, 1);
    k_add<<<(NX+255)/256, 256, 0, stream>>>(x, tmp, NX);
  }
}
